// RWKVBlock_36490042146954
// MI455X (gfx1250) — compile-verified, matrix operand order checked by reading, not run on hardware
//
#include <hip/hip_runtime.h>
#include <hip/hip_bf16.h>

// ---------------------------------------------------------------------------
// Types for gfx1250 WMMA (wave32): 16x16x32 bf16 -> f32 accumulate
// ---------------------------------------------------------------------------
typedef __attribute__((ext_vector_type(16))) __bf16 v16bf;
typedef __attribute__((ext_vector_type(8)))  float  v8f;
typedef int v4i __attribute__((vector_size(16)));   // matches async-LDS builtin param

union FragBF16 {
    uint4 u4[2];   // 32 bytes
    v16bf v;       // 16 x bf16
};

#define AS1 __attribute__((address_space(1)))
#define AS3 __attribute__((address_space(3)))

#if __has_builtin(__builtin_amdgcn_global_load_async_to_lds_b128)
#define USE_ASYNC_LDS 1
#else
#define USE_ASYNC_LDS 0
#endif

#if __has_builtin(__builtin_amdgcn_s_wait_asynccnt)
#define WAIT_ASYNC(n) __builtin_amdgcn_s_wait_asynccnt(n)
#else
#define WAIT_ASYNC(n) asm volatile("s_wait_asynccnt %0" :: "i"(n) : "memory")
#endif

#if USE_ASYNC_LDS
__device__ __forceinline__ void async_b128(const void* g, void* l) {
    __builtin_amdgcn_global_load_async_to_lds_b128(
        (AS1 v4i*)g, (AS3 v4i*)l, 0, 0);
}
#endif

__device__ __forceinline__ unsigned short f2bf(float f) {
    unsigned int u = __float_as_uint(f);
    u += 0x7FFFu + ((u >> 16) & 1u);   // round-to-nearest-even
    return (unsigned short)(u >> 16);
}

// ---------------------------------------------------------------------------
// Pack weight [K,N] f32 -> transposed [N,K] bf16 (one-time; amortized over M)
// ---------------------------------------------------------------------------
__global__ void pack_wT_kernel(const float* __restrict__ W,
                               unsigned short* __restrict__ WT,
                               int K, int N) {
    size_t id = (size_t)blockIdx.x * 256 + threadIdx.x;
    size_t total = (size_t)K * N;
    if (id >= total) return;
    int k = (int)(id / N);
    int n = (int)(id % N);
    WT[(size_t)n * K + k] = f2bf(W[id]);
}

// ---------------------------------------------------------------------------
// LayerNorm + token-shift + mix (time-mixing) -> bf16 xk/xv/xr
// one block (256 threads) per token; D = 1024 -> 4 elems/thread
// ---------------------------------------------------------------------------
__global__ __launch_bounds__(256)
void ln_mix_time_kernel(const float* __restrict__ x,
                        const float* __restrict__ g, const float* __restrict__ bta,
                        const float* __restrict__ mixk, const float* __restrict__ mixv,
                        const float* __restrict__ mixr,
                        unsigned short* __restrict__ xk,
                        unsigned short* __restrict__ xv,
                        unsigned short* __restrict__ xr,
                        int L, int D) {
    int token = blockIdx.x;
    int t = token % L;
    bool hasPrev = (t > 0);
    const float* cur  = x + (size_t)token * D;
    const float* prev = cur - D;
    int tid = threadIdx.x;

    float c[4], pv[4];
    float4 p = make_float4(0.f, 0.f, 0.f, 0.f);
    #pragma unroll
    for (int j = 0; j < 4; j++) {
        int d = tid + j * 256;
        float xc = cur[d];
        float xp = hasPrev ? prev[d] : 0.f;
        c[j] = xc; pv[j] = xp;
        p.x += xc; p.y += xc * xc; p.z += xp; p.w += xp * xp;
    }
    __shared__ float4 red[256];
    red[tid] = p;
    __syncthreads();
    for (int s = 128; s > 0; s >>= 1) {
        if (tid < s) {
            float4 a = red[tid], b = red[tid + s];
            a.x += b.x; a.y += b.y; a.z += b.z; a.w += b.w;
            red[tid] = a;
        }
        __syncthreads();
    }
    float4 tot = red[0];
    float invD = 1.0f / (float)D;
    float m_c = tot.x * invD, v_c = tot.y * invD - m_c * m_c;
    float m_p = tot.z * invD, v_p = tot.w * invD - m_p * m_p;
    float rs_c = rsqrtf(v_c + 1e-5f);
    float rs_p = rsqrtf(v_p + 1e-5f);

    #pragma unroll
    for (int j = 0; j < 4; j++) {
        int d = tid + j * 256;
        float gn = g[d], bn = bta[d];
        float xn = (c[j] - m_c) * rs_c * gn + bn;
        float xp = hasPrev ? ((pv[j] - m_p) * rs_p * gn + bn) : 0.f;
        float mk = mixk[d], mv = mixv[d], mr = mixr[d];
        size_t o = (size_t)token * D + d;
        xk[o] = f2bf(xn * mk + xp * (1.f - mk));
        xv[o] = f2bf(xn * mv + xp * (1.f - mv));
        xr[o] = f2bf(xn * mr + xp * (1.f - mr));
    }
}

// ---------------------------------------------------------------------------
// LayerNorm + 0.5 token-shift mix (channel-mixing) -> bf16 ym (mixed), yn
// ---------------------------------------------------------------------------
__global__ __launch_bounds__(256)
void ln_mix_chan_kernel(const float* __restrict__ x,
                        const float* __restrict__ g, const float* __restrict__ bta,
                        unsigned short* __restrict__ ym,
                        unsigned short* __restrict__ yn_out,
                        int L, int D) {
    int token = blockIdx.x;
    int t = token % L;
    bool hasPrev = (t > 0);
    const float* cur  = x + (size_t)token * D;
    const float* prev = cur - D;
    int tid = threadIdx.x;

    float c[4], pv[4];
    float4 p = make_float4(0.f, 0.f, 0.f, 0.f);
    #pragma unroll
    for (int j = 0; j < 4; j++) {
        int d = tid + j * 256;
        float xc = cur[d];
        float xp = hasPrev ? prev[d] : 0.f;
        c[j] = xc; pv[j] = xp;
        p.x += xc; p.y += xc * xc; p.z += xp; p.w += xp * xp;
    }
    __shared__ float4 red[256];
    red[tid] = p;
    __syncthreads();
    for (int s = 128; s > 0; s >>= 1) {
        if (tid < s) {
            float4 a = red[tid], b = red[tid + s];
            a.x += b.x; a.y += b.y; a.z += b.z; a.w += b.w;
            red[tid] = a;
        }
        __syncthreads();
    }
    float4 tot = red[0];
    float invD = 1.0f / (float)D;
    float m_c = tot.x * invD, v_c = tot.y * invD - m_c * m_c;
    float m_p = tot.z * invD, v_p = tot.w * invD - m_p * m_p;
    float rs_c = rsqrtf(v_c + 1e-5f);
    float rs_p = rsqrtf(v_p + 1e-5f);

    #pragma unroll
    for (int j = 0; j < 4; j++) {
        int d = tid + j * 256;
        float gn = g[d], bn = bta[d];
        float yn = (c[j] - m_c) * rs_c * gn + bn;
        float yp = hasPrev ? ((pv[j] - m_p) * rs_p * gn + bn) : 0.f;
        size_t o = (size_t)token * D + d;
        ym[o]     = f2bf(0.5f * yn + 0.5f * yp);
        yn_out[o] = f2bf(yn);
    }
}

// ---------------------------------------------------------------------------
// WKV sequential scan over L, fused with r * wkv -> bf16 output for Wo GEMM.
// One thread per (b, d) channel; lanes within a wave cover consecutive d
// so k/v/r loads are fully coalesced each step.
// ---------------------------------------------------------------------------
__global__ __launch_bounds__(256)
void wkv_kernel(const float* __restrict__ kf, const float* __restrict__ vf,
                const float* __restrict__ rf,
                const float* __restrict__ td, const float* __restrict__ tf,
                unsigned short* __restrict__ rwkv,
                int L, int D) {
    int id = blockIdx.x * 256 + threadIdx.x;   // < B*D
    int d = id % D;
    int b = id / D;
    float ew = __expf(-__expf(td[d]));
    float u  = tf[d];
    size_t base = (size_t)b * L * D + d;

    // t = 0 (u-boosted first term)
    float k0 = kf[base], v0 = vf[base];
    float e0 = __expf(u + k0);
    float a  = e0 * v0;
    float bb = e0;
    rwkv[base] = f2bf(rf[base] * (a / (bb + 1e-8f)));

    for (int t = 1; t < L; t++) {
        size_t o = base + (size_t)t * D;
        float kt = kf[o], vt = vf[o];
        float e = __expf(kt);
        a  = ew * a  + e * vt;
        bb = ew * bb + e;
        rwkv[o] = f2bf(rf[o] * (a / (bb + 1e-8f)));
    }
}

// ---------------------------------------------------------------------------
// bf16 WMMA GEMM: C[M,N] = epilogue( A[M,K](bf16) @ WT[N,K](bf16)^T )
//   128x128 tile / block, 256 threads = 8 wave32 (2x4), BK = 32.
//   Each wave: 4x2 tiles of v_wmma_f32_16x16x32_bf16.
//   Staging: double-buffered GLOBAL_LOAD_ASYNC_TO_LDS_B128 (ASYNCcnt
//   pipeline) when available; synchronous b128+ds_store fallback otherwise.
// Epilogue MODE: 0 = f32, 1 = f32 + aux0 (residual), 2 = sigmoid f32,
//                3 = relu^2 -> bf16, 4 = aux0 + aux1 * acc (final output)
// ---------------------------------------------------------------------------
template<int MODE>
__global__ __launch_bounds__(256)
void gemm_bf16_wmma(const unsigned short* __restrict__ A,
                    const unsigned short* __restrict__ WT,
                    const float* __restrict__ aux0,
                    const float* __restrict__ aux1,
                    float* __restrict__ outF,
                    unsigned short* __restrict__ outH,
                    int M, int N, int K) {
    __shared__ unsigned short As[2][128 * 32];
    __shared__ unsigned short Bs[2][128 * 32];

    int tid = threadIdx.x;
    int m0 = blockIdx.y * 128;
    int n0 = blockIdx.x * 128;
    int w = tid >> 5, l = tid & 31;
    int waveM = w >> 2;       // 0..1 (64 rows each)
    int waveN = w & 3;        // 0..3 (32 cols each)
    int half = l >> 4;        // half-wave select for K split
    int lr = l & 15;          // row/col within 16

    v8f zero = {0.f, 0.f, 0.f, 0.f, 0.f, 0.f, 0.f, 0.f};
    v8f acc[4][2];
    #pragma unroll
    for (int i = 0; i < 4; i++)
        #pragma unroll
        for (int j = 0; j < 2; j++) acc[i][j] = zero;

    int idx0 = tid, idx1 = tid + 256;
    int r0 = idx0 >> 2, c0 = idx0 & 3;
    int r1 = idx1 >> 2, c1 = idx1 & 3;

#if USE_ASYNC_LDS
    // ---- async (ASYNCcnt) double-buffered pipeline -------------------------
    auto stage = [&](int buf, int k0) {
        // 4 async b128 copies per thread; data never touches VGPRs
        async_b128(A  + (size_t)(m0 + r0) * K + k0 + c0 * 8, &As[buf][(size_t)idx0 * 8]);
        async_b128(A  + (size_t)(m0 + r1) * K + k0 + c1 * 8, &As[buf][(size_t)idx1 * 8]);
        async_b128(WT + (size_t)(n0 + r0) * K + k0 + c0 * 8, &Bs[buf][(size_t)idx0 * 8]);
        async_b128(WT + (size_t)(n0 + r1) * K + k0 + c1 * 8, &Bs[buf][(size_t)idx1 * 8]);
    };

    stage(0, 0);
    int buf = 0;
    for (int k0 = 0; k0 < K; k0 += 32) {
        if (k0 + 32 < K) {
            stage(buf ^ 1, k0 + 32);   // prefetch next tile into other buffer
            WAIT_ASYNC(4);             // in-order: current tile's 4 ops done
        } else {
            WAIT_ASYNC(0);
        }
        __syncthreads();

        FragBF16 af[4], bfr[2];
        #pragma unroll
        for (int mi = 0; mi < 4; mi++) {
            int row = waveM * 64 + mi * 16 + lr;
            af[mi].u4[0] = *(const uint4*)(&As[buf][row * 32 + half * 8]);
            af[mi].u4[1] = *(const uint4*)(&As[buf][row * 32 + half * 8 + 16]);
        }
        #pragma unroll
        for (int ni = 0; ni < 2; ni++) {
            int col = waveN * 32 + ni * 16 + lr;
            bfr[ni].u4[0] = *(const uint4*)(&Bs[buf][col * 32 + half * 8]);
            bfr[ni].u4[1] = *(const uint4*)(&Bs[buf][col * 32 + half * 8 + 16]);
        }

        #pragma unroll
        for (int mi = 0; mi < 4; mi++)
            #pragma unroll
            for (int ni = 0; ni < 2; ni++)
                acc[mi][ni] = __builtin_amdgcn_wmma_f32_16x16x32_bf16(
                    false, af[mi].v, false, bfr[ni].v,
                    (short)0, acc[mi][ni], false, false);
        __syncthreads();   // all reads of buf done before it is re-staged
        buf ^= 1;
    }
#else
    // ---- synchronous fallback ---------------------------------------------
    for (int k0 = 0; k0 < K; k0 += 32) {
        ((uint4*)As[0])[idx0] = *(const uint4*)(A  + (size_t)(m0 + r0) * K + k0 + c0 * 8);
        ((uint4*)As[0])[idx1] = *(const uint4*)(A  + (size_t)(m0 + r1) * K + k0 + c1 * 8);
        ((uint4*)Bs[0])[idx0] = *(const uint4*)(WT + (size_t)(n0 + r0) * K + k0 + c0 * 8);
        ((uint4*)Bs[0])[idx1] = *(const uint4*)(WT + (size_t)(n0 + r1) * K + k0 + c1 * 8);
        __syncthreads();
        FragBF16 af[4], bfr[2];
        #pragma unroll
        for (int mi = 0; mi < 4; mi++) {
            int row = waveM * 64 + mi * 16 + lr;
            af[mi].u4[0] = *(const uint4*)(&As[0][row * 32 + half * 8]);
            af[mi].u4[1] = *(const uint4*)(&As[0][row * 32 + half * 8 + 16]);
        }
        #pragma unroll
        for (int ni = 0; ni < 2; ni++) {
            int col = waveN * 32 + ni * 16 + lr;
            bfr[ni].u4[0] = *(const uint4*)(&Bs[0][col * 32 + half * 8]);
            bfr[ni].u4[1] = *(const uint4*)(&Bs[0][col * 32 + half * 8 + 16]);
        }
        #pragma unroll
        for (int mi = 0; mi < 4; mi++)
            #pragma unroll
            for (int ni = 0; ni < 2; ni++)
                acc[mi][ni] = __builtin_amdgcn_wmma_f32_16x16x32_bf16(
                    false, af[mi].v, false, bfr[ni].v,
                    (short)0, acc[mi][ni], false, false);
        __syncthreads();
    }
#endif

    // epilogue: VGPR r of 16x16 tile -> row = r + 8*half, col = lr
    #pragma unroll
    for (int mi = 0; mi < 4; mi++) {
        #pragma unroll
        for (int ni = 0; ni < 2; ni++) {
            int colg = n0 + waveN * 32 + ni * 16 + lr;
            #pragma unroll
            for (int r = 0; r < 8; r++) {
                int rowg = m0 + waveM * 64 + mi * 16 + half * 8 + r;
                size_t o = (size_t)rowg * N + colg;
                float vv = acc[mi][ni][r];
                if (MODE == 0) {
                    outF[o] = vv;
                } else if (MODE == 1) {
                    outF[o] = vv + aux0[o];
                } else if (MODE == 2) {
                    outF[o] = 1.0f / (1.0f + __expf(-vv));
                } else if (MODE == 3) {
                    float t = vv > 0.f ? vv : 0.f;
                    outH[o] = f2bf(t * t);
                } else { // MODE == 4
                    outF[o] = aux0[o] + aux1[o] * vv;
                }
            }
        }
    }
}

// ---------------------------------------------------------------------------
// Host orchestration
// ---------------------------------------------------------------------------
extern "C" void kernel_launch(void* const* d_in, const int* in_sizes, int n_in,
                              void* d_out, int out_size, void* d_ws, size_t ws_size,
                              hipStream_t stream) {
    (void)in_sizes; (void)n_in; (void)out_size; (void)ws_size;
    const int Bn = 8, L = 2048, D = 1024, F = 4096;
    const int M = Bn * L;   // 16384

    const float* x      = (const float*)d_in[0];
    const float* ln1_g  = (const float*)d_in[1];
    const float* ln1_b  = (const float*)d_in[2];
    const float* Wk     = (const float*)d_in[3];
    const float* Wv     = (const float*)d_in[4];
    const float* Wr     = (const float*)d_in[5];
    const float* Wo     = (const float*)d_in[6];
    const float* tdec   = (const float*)d_in[7];
    const float* tfirst = (const float*)d_in[8];
    const float* mix_k  = (const float*)d_in[9];
    const float* mix_v  = (const float*)d_in[10];
    const float* mix_r  = (const float*)d_in[11];
    const float* ln2_g  = (const float*)d_in[12];
    const float* ln2_b  = (const float*)d_in[13];
    const float* Wfk    = (const float*)d_in[14];
    const float* Wfv    = (const float*)d_in[15];
    const float* Wfr    = (const float*)d_in[16];

    char* ws = (char*)d_ws;
    size_t off = 0;
    auto alloc = [&](size_t bytes) -> char* {
        off = (off + 255) & ~(size_t)255;
        char* p = ws + off;
        off += bytes;
        return p;
    };

    // bf16 transposed weights [N,K]
    unsigned short* wkT  = (unsigned short*)alloc((size_t)D * D * 2);
    unsigned short* wvT  = (unsigned short*)alloc((size_t)D * D * 2);
    unsigned short* wrT  = (unsigned short*)alloc((size_t)D * D * 2);
    unsigned short* woT  = (unsigned short*)alloc((size_t)D * D * 2);
    unsigned short* wfkT = (unsigned short*)alloc((size_t)D * F * 2); // N=F,K=D
    unsigned short* wfvT = (unsigned short*)alloc((size_t)F * D * 2); // N=D,K=F
    unsigned short* wfrT = (unsigned short*)alloc((size_t)D * D * 2);

    // activations
    unsigned short* xk_bf = (unsigned short*)alloc((size_t)M * D * 2);
    unsigned short* xv_bf = (unsigned short*)alloc((size_t)M * D * 2);
    unsigned short* xr_bf = (unsigned short*)alloc((size_t)M * D * 2);
    float* kf = (float*)alloc((size_t)M * D * 4);
    float* vf = (float*)alloc((size_t)M * D * 4);   // contiguous with kf
    float* rf = (float*)alloc((size_t)M * D * 4);
    float* rr = (float*)alloc((size_t)M * D * 4);

    // reuse (lifetimes disjoint):
    unsigned short* rwkv_bf = xr_bf;                 // after r GEMM consumes xr
    float*          x1      = rf;                    // after wkv consumes r
    unsigned short* ym_bf   = xk_bf;                 // after k GEMM
    unsigned short* yn_bf   = xv_bf;                 // after v GEMM
    unsigned short* kk_bf   = (unsigned short*)kf;   // spans kf+vf (128 MB)

    dim3 blk(256);

    // 1) pack weights -> bf16 transposed
    {
        int nDD = (D * D + 255) / 256;
        int nDF = (D * F + 255) / 256;
        pack_wT_kernel<<<nDD, blk, 0, stream>>>(Wk,  wkT,  D, D);
        pack_wT_kernel<<<nDD, blk, 0, stream>>>(Wv,  wvT,  D, D);
        pack_wT_kernel<<<nDD, blk, 0, stream>>>(Wr,  wrT,  D, D);
        pack_wT_kernel<<<nDD, blk, 0, stream>>>(Wo,  woT,  D, D);
        pack_wT_kernel<<<nDF, blk, 0, stream>>>(Wfk, wfkT, D, F);
        pack_wT_kernel<<<nDF, blk, 0, stream>>>(Wfv, wfvT, F, D);
        pack_wT_kernel<<<nDD, blk, 0, stream>>>(Wfr, wfrT, D, D);
    }

    // 2) LN1 + shift + mix
    ln_mix_time_kernel<<<M, blk, 0, stream>>>(x, ln1_g, ln1_b, mix_k, mix_v, mix_r,
                                              xk_bf, xv_bf, xr_bf, L, D);

    // 3-5) k, v, r GEMMs (M x 1024 x 1024)
    dim3 gDD(D / 128, M / 128);
    gemm_bf16_wmma<0><<<gDD, blk, 0, stream>>>(xk_bf, wkT, nullptr, nullptr, kf, nullptr, M, D, D);
    gemm_bf16_wmma<0><<<gDD, blk, 0, stream>>>(xv_bf, wvT, nullptr, nullptr, vf, nullptr, M, D, D);
    gemm_bf16_wmma<2><<<gDD, blk, 0, stream>>>(xr_bf, wrT, nullptr, nullptr, rf, nullptr, M, D, D);

    // 6) WKV scan fused with r*wkv -> bf16
    wkv_kernel<<<(Bn * D) / 256, blk, 0, stream>>>(kf, vf, rf, tdec, tfirst, rwkv_bf, L, D);

    // 7) x1 = x + (r*wkv) @ Wo
    gemm_bf16_wmma<1><<<gDD, blk, 0, stream>>>(rwkv_bf, woT, x, nullptr, x1, nullptr, M, D, D);

    // 8) LN2 + 0.5 shift-mix
    ln_mix_chan_kernel<<<M, blk, 0, stream>>>(x1, ln2_g, ln2_b, ym_bf, yn_bf, L, D);

    // 9) kk = relu(ym @ Wfk)^2 -> bf16   (M x 4096 x 1024)
    dim3 gDF(F / 128, M / 128);
    gemm_bf16_wmma<3><<<gDF, blk, 0, stream>>>(ym_bf, wfkT, nullptr, nullptr, nullptr, kk_bf, M, F, D);

    // 10) rr = sigmoid(yn @ Wfr)
    gemm_bf16_wmma<2><<<gDD, blk, 0, stream>>>(yn_bf, wfrT, nullptr, nullptr, rr, nullptr, M, D, D);

    // 11) out = x1 + rr * (kk @ Wfv)     (M x 1024 x 4096)
    gemm_bf16_wmma<4><<<gDD, blk, 0, stream>>>(kk_bf, wfvT, x1, rr, (float*)d_out, nullptr, M, D, F);
}
